// Net_67027259622057
// MI455X (gfx1250) — matrix operand order repaired and checked by reading, not compiled, not run on hardware
//
#include <hip/hip_runtime.h>
#include <hip/hip_bf16.h>

typedef __attribute__((ext_vector_type(16))) _Float16 v16h;
typedef __attribute__((ext_vector_type(8)))  _Float16 v8h;
typedef __attribute__((ext_vector_type(8)))  float    v8f;

#define D_IN 128

// ---------------------------------------------------------------------------
// zero a float buffer
// ---------------------------------------------------------------------------
__global__ void zero_f32(float* __restrict__ p, long long n) {
    long long i = (long long)blockIdx.x * blockDim.x + threadIdx.x;
    if (i < n) p[i] = 0.0f;
}

// ---------------------------------------------------------------------------
// deg[dst[e]] += 1  (deg pre-zeroed)
// ---------------------------------------------------------------------------
__global__ void degree_kernel(const long long* __restrict__ dst,
                              float* __restrict__ deg, long long E) {
    long long e = (long long)blockIdx.x * blockDim.x + threadIdx.x;
    if (e < E) atomicAdd(&deg[dst[e]], 1.0f);
}

// dinv[i] = rsqrt(deg[i] + 1)  (self loop)
__global__ void rsqrt_kernel(float* __restrict__ d, int n) {
    int i = blockIdx.x * blockDim.x + threadIdx.x;
    if (i < n) d[i] = __frsqrt_rn(d[i] + 1.0f);
}

// ---------------------------------------------------------------------------
// f32 -> f16 elementwise
// ---------------------------------------------------------------------------
__global__ void cvt_f16(const float* __restrict__ in, _Float16* __restrict__ out,
                        long long n) {
    long long i = (long long)blockIdx.x * blockDim.x + threadIdx.x;
    if (i < n) out[i] = (_Float16)in[i];
}

// Wt[n][k] = (f16) W[k][n]   (W is 128 x COLS row-major)
template <int COLS>
__global__ void transpose_w(const float* __restrict__ W, _Float16* __restrict__ Wt) {
    int i = blockIdx.x * blockDim.x + threadIdx.x;   // i = n*128 + k
    if (i >= COLS * D_IN) return;
    int n = i >> 7, k = i & 127;
    Wt[i] = (_Float16)W[(size_t)k * COLS + n];
}

// ---------------------------------------------------------------------------
// WMMA GEMM:  Y[nrows x COLS](f32) = Xh[nrows x 128](f16) * Wt^T  (Wt: COLS x 128 f16)
// One wave per 16-row tile. Fragments are pure b128 loads:
//   A (16x32 f16): lane L: M=L%16, h=L/16; elems 0..7  = halfs k0+8h  .. +7
//                                          elems 8..15 = halfs k0+16+8h .. +7
//   B (32x16 f16): lane L: N=L%16, h=L/16; elems 0..7  = halfs k0+8h  .. +7
//                                          elems 8..15 = halfs k0+16+8h .. +7
//   C/D (16x16 f32): lane L: N=L%16; VGPR j: M = j + 8h
// ---------------------------------------------------------------------------
template <int COLS>
__global__ void gemm_wmma(const _Float16* __restrict__ Xh,
                          const _Float16* __restrict__ Wt,
                          float* __restrict__ Y, int nRowTiles) {
    const int lane    = threadIdx.x & 31;
    const int wave    = threadIdx.x >> 5;
    const int rowTile = blockIdx.x * (blockDim.x >> 5) + wave;
    if (rowTile >= nRowTiles) return;   // wave-uniform: EXEC stays all-ones

    const int rowBase = rowTile * 16;
    const int m    = lane & 15;
    const int half = lane >> 4;

    const _Float16* xrow = Xh + (size_t)(rowBase + m) * D_IN;

    // A fragments for the whole K=128 strip (4 x v16h = 32 VGPRs)
    v16h A[4];
#pragma unroll
    for (int t = 0; t < 4; ++t) {
        const v8h lo = *(const v8h*)(xrow + 32 * t + 8 * half);
        const v8h hi = *(const v8h*)(xrow + 32 * t + 16 + 8 * half);
#pragma unroll
        for (int i = 0; i < 8; ++i) { A[t][i] = lo[i]; A[t][8 + i] = hi[i]; }
    }

    for (int nt = 0; nt < COLS / 16; ++nt) {
        const int colBase = nt * 16;
        const _Float16* wcol = Wt + (size_t)(colBase + m) * D_IN;
        v8f acc = {};
#pragma unroll
        for (int t = 0; t < 4; ++t) {
            v16h b;
            const v8h blo = *(const v8h*)(wcol + 32 * t + 8 * half);
            const v8h bhi = *(const v8h*)(wcol + 32 * t + 16 + 8 * half);
#pragma unroll
            for (int i = 0; i < 8; ++i) { b[i] = blo[i]; b[8 + i] = bhi[i]; }
            acc = __builtin_amdgcn_wmma_f32_16x16x32_f16(
                false, A[t], false, b, (short)0, acc, false, false);
        }
#pragma unroll
        for (int j = 0; j < 8; ++j) {
            Y[(size_t)(rowBase + j + 8 * half) * COLS + colBase + m] = acc[j];
        }
    }
}

// ---------------------------------------------------------------------------
// Edge scatter: agg[dst] += hW[src] * (dinv[src]*dinv[dst])
// One wave per edge; one b128 (COLS=128) / b64 (COLS=64) load per lane,
// then per-dword global_atomic_add_f32 (agg + h both resident in 192MB L2).
// ---------------------------------------------------------------------------
template <int COLS>
__global__ void scatter_edges(const long long* __restrict__ src,
                              const long long* __restrict__ dst,
                              const float* __restrict__ hW,
                              const float* __restrict__ dinv,
                              float* __restrict__ agg, long long E) {
    const int lane = threadIdx.x & 31;
    const long long e =
        (long long)blockIdx.x * (blockDim.x >> 5) + (threadIdx.x >> 5);
    if (e >= E) return;

    const long long s = src[e];
    const long long d = dst[e];
    const float nrm = dinv[s] * dinv[d];

    const float* hs = hW + s * COLS;
    float*       ad = agg + d * COLS;

    if constexpr (COLS == 128) {
        const float4 hv = ((const float4*)hs)[lane];
        const int c = 4 * lane;
        atomicAdd(&ad[c + 0], hv.x * nrm);
        atomicAdd(&ad[c + 1], hv.y * nrm);
        atomicAdd(&ad[c + 2], hv.z * nrm);
        atomicAdd(&ad[c + 3], hv.w * nrm);
    } else {
        const float2 hv = ((const float2*)hs)[lane];
        const int c = 2 * lane;
        atomicAdd(&ad[c + 0], hv.x * nrm);
        atomicAdd(&ad[c + 1], hv.y * nrm);
    }
}

// ---------------------------------------------------------------------------
// out[i,c] = (relu?)( agg[i,c] + hW[i,c]*dinv[i]^2 + bias[c] )
// OUT_T = _Float16 lets layer 1 emit the next GEMM's A operand directly.
// ---------------------------------------------------------------------------
template <int COLS, bool RELU, typename OUT_T>
__global__ void finish_kernel(const float* __restrict__ agg,
                              const float* __restrict__ hW,
                              const float* __restrict__ dinv,
                              const float* __restrict__ bias,
                              OUT_T* __restrict__ out, long long total) {
    long long i = (long long)blockIdx.x * blockDim.x + threadIdx.x;
    if (i >= total) return;
    const int node = (int)(i / COLS);
    const int c    = (int)(i % COLS);
    const float di = dinv[node];
    float v = agg[i] + hW[i] * (di * di) + bias[c];
    if (RELU) v = fmaxf(v, 0.0f);
    out[i] = (OUT_T)v;
}

// ---------------------------------------------------------------------------
// Launcher
// ---------------------------------------------------------------------------
extern "C" void kernel_launch(void* const* d_in, const int* in_sizes, int n_in,
                              void* d_out, int out_size, void* d_ws, size_t ws_size,
                              hipStream_t stream) {
    const float*     x   = (const float*)d_in[0];
    const long long* ei  = (const long long*)d_in[1];
    const float*     W1  = (const float*)d_in[2];
    const float*     b1  = (const float*)d_in[3];
    const float*     W2  = (const float*)d_in[4];
    const float*     b2  = (const float*)d_in[5];
    float*           out = (float*)d_out;

    const int       N = in_sizes[0] / D_IN;      // 100000
    const long long E = in_sizes[1] / 2;         // 1600000
    const long long* src = ei;
    const long long* dst = ei + E;

    // Workspace layout
    char*  ws = (char*)d_ws;
    size_t off = 0;
    float*    dinv   = (float*)(ws + off);    off += ((size_t)N * 4 + 255) & ~(size_t)255;
    float*    hW     = (float*)(ws + off);    off += (size_t)N * 128 * 4;  // layer2 uses N*64
    float*    agg1   = (float*)(ws + off);    off += (size_t)N * 128 * 4;
    _Float16* feat16 = (_Float16*)(ws + off); off += (size_t)N * 128 * 2;  // x(f16), then h1(f16)
    _Float16* W1t    = (_Float16*)(ws + off); off += (size_t)128 * 128 * 2;
    _Float16* W2t    = (_Float16*)(ws + off);
    (void)ws_size; (void)n_in; (void)out_size;

    const int BLK = 256;
    const int wavesPerBlk = BLK / 32;
    const int nRowTiles = N / 16;                // 6250 (exact)
    const int gemmBlocks = (nRowTiles + wavesPerBlk - 1) / wavesPerBlk;
    const int edgeBlocks = (int)((E + wavesPerBlk - 1) / wavesPerBlk);

    // ---- degrees -> dinv -------------------------------------------------
    zero_f32<<<(N + BLK - 1) / BLK, BLK, 0, stream>>>(dinv, N);
    degree_kernel<<<(int)((E + BLK - 1) / BLK), BLK, 0, stream>>>(dst, dinv, E);
    rsqrt_kernel<<<(N + BLK - 1) / BLK, BLK, 0, stream>>>(dinv, N);

    // ---- operand precision prep ------------------------------------------
    {
        const long long nx = (long long)N * 128;
        cvt_f16<<<(int)((nx + BLK - 1) / BLK), BLK, 0, stream>>>(x, feat16, nx);
        transpose_w<128><<<(128 * 128 + BLK - 1) / BLK, BLK, 0, stream>>>(W1, W1t);
        transpose_w<64><<<(64 * 128 + BLK - 1) / BLK, BLK, 0, stream>>>(W2, W2t);
    }

    // ---- layer 1: h1 = relu(Ahat @ (x W1) + b1)  (h1 emitted as f16) -----
    {
        gemm_wmma<128><<<gemmBlocks, BLK, 0, stream>>>(feat16, W1t, hW, nRowTiles);

        const long long tot = (long long)N * 128;
        zero_f32<<<(int)((tot + BLK - 1) / BLK), BLK, 0, stream>>>(agg1, tot);
        scatter_edges<128><<<edgeBlocks, BLK, 0, stream>>>(src, dst, hW, dinv, agg1, E);
        // x(f16) is dead after GEMM1: overwrite feat16 with h1(f16)
        finish_kernel<128, true, _Float16><<<(int)((tot + BLK - 1) / BLK), BLK, 0, stream>>>(
            agg1, hW, dinv, b1, feat16, tot);
    }

    // ---- layer 2: out = Ahat @ (h1 W2) + b2 ------------------------------
    {
        gemm_wmma<64><<<gemmBlocks, BLK, 0, stream>>>(feat16, W2t, hW, nRowTiles);

        const long long tot = (long long)N * 64;
        zero_f32<<<(int)((tot + BLK - 1) / BLK), BLK, 0, stream>>>(out, tot);
        scatter_edges<64><<<edgeBlocks, BLK, 0, stream>>>(src, dst, hW, dinv, out, E);
        finish_kernel<64, false, float><<<(int)((tot + BLK - 1) / BLK), BLK, 0, stream>>>(
            out, hW, dinv, b2, out, tot);
    }
}
